// PureGraphEncoder_24790551233228
// MI455X (gfx1250) — compile-verified, matrix operand order checked by reading, not run on hardware
//
#include <hip/hip_runtime.h>

#define N_DOCS   4000
#define N_WORDS  26000
#define DOC_DIM  10000
#define WORD_DIM 300
#define OUT_DIM  768
#define NNODES   (N_DOCS + N_WORDS)
#define NEDGES   262144
#define NMASK    4096

// padded K slots (multiples of 32, zero-filled)
#define DOC_DIM_P  10016
#define WORD_DIM_P 320
#define K_CAT1     (2 * WORD_DIM_P)   // [agg1 | x]        = 640
#define K_CAT2     (2 * OUT_DIM)      // [agg2 | h1]       = 1536

typedef __attribute__((ext_vector_type(16))) __bf16 v16bf;
typedef __attribute__((ext_vector_type(8)))  __bf16 v8bf;
typedef __attribute__((ext_vector_type(4)))  __bf16 v4bf;
typedef __attribute__((ext_vector_type(8)))  float  v8f;

// fp32 -> bf16, round-to-nearest-even (only in the bandwidth-bound cast pass)
__device__ inline __bf16 f2bf(float f) {
    union { float f; unsigned u; } in; in.f = f;
    unsigned u = in.u + 0x7FFFu + ((in.u >> 16) & 1u);
    union { unsigned short s; __bf16 b; } out; out.s = (unsigned short)(u >> 16);
    return out.b;
}
__device__ inline v4bf cvt4(float4 v) {
    v4bf r; r[0] = f2bf(v.x); r[1] = f2bf(v.y); r[2] = f2bf(v.z); r[3] = f2bf(v.w);
    return r;
}

// Inverse of the wave32 16-bit WMMA fragment layout (ISA 7.12.2, 16x32 tile):
// element (r, k) -> lane = (r&15) + 16*((k>>3)&1), elem = (k&7) + 8*((k>>4)&1).
// Stored [subtile][lane][elem]: a lane's fragment = 32 contiguous bytes
// (2x ds_load_b128); an 8-aligned k-group = 8 contiguous bf16 (1x ds_store_b128).
__device__ inline int frag_index(int r, int k) {
    int t    = r >> 4;
    int lane = (r & 15) + (((k >> 3) & 1) << 4);
    int e    = (k & 7) + (((k >> 4) & 1) << 3);
    return ((t << 5) + lane) * 16 + e;
}

__device__ inline v16bf ld_frag(const __bf16* p) {
    return *reinterpret_cast<const v16bf*>(p);
}

// cast fp32 [rows x K] -> bf16 [rows x Kp] slot inside a row of stride ldout,
// zero padding K..Kp-1 (K%4==0, Kp%4==0). Enables K-concatenated operands.
__global__ __launch_bounds__(256)
void cast_pad_bf16(const float* __restrict__ in, __bf16* __restrict__ out,
                   int rows, int K, int Kp, int ldout)
{
    const int gp = Kp >> 2;
    long idx = (long)blockIdx.x * blockDim.x + threadIdx.x;
    if (idx >= (long)rows * gp) return;
    int  g = (int)(idx % gp);
    long r = idx / gp;
    int  k = g * 4;
    v4bf v = {};
    if (k < K) {
        float4 f = *reinterpret_cast<const float4*>(in + r * K + k);
        v = cvt4(f);
    }
    *reinterpret_cast<v4bf*>(out + r * (long)ldout + k) = v;
}

// C[M,N] = A[M,Kp] @ W[N,Kp]^T (+ bias) (ReLU optional), A rows via rowmap.
// Operands pre-cast bf16, Kp % 32 == 0 (zero padded) -> no K guards; OOB
// rows/cols address-clamped only (row m of D depends only on row m of A).
// Block = 256 threads = 8 wave32s; block tile 128x64; wave tile 32x32:
// 4 fragments (8 ds_load_b128) feed 4 WMMAs per K-step with full reg reuse.
// Double-buffered LDS (24 KB), register staging, one barrier per K-step.
template<bool RELU>
__global__ __launch_bounds__(256)
void gemm_bf16_wmma(const __bf16* __restrict__ A, const __bf16* __restrict__ W,
                    const float* __restrict__ bias, const int* __restrict__ rowmap,
                    float* __restrict__ C, int M, int N, int Kp)
{
    constexpr int BM = 128, BN = 64, BK = 32;
    __shared__ __align__(32) __bf16 sA[2][BM * BK];   // 8 m-subtiles, frag layout
    __shared__ __align__(32) __bf16 sB[2][BK * BN];   // 4 n-subtiles, frag layout

    const int tid  = threadIdx.x;
    const int lane = tid & 31;
    const int wave = tid >> 5;          // 0..7
    const int wm   = wave >> 1;         // 0..3 : pair of m-subtiles
    const int wn   = wave & 1;          // 0..1 : pair of n-subtiles
    const int blockM = blockIdx.y * BM;
    const int blockN = blockIdx.x * BN;

    // --- staging geometry (branch-free, hoisted) ---
    // A: two 8-wide k-group slots per thread; B: one.
    int dstA[2]; long aoff[2];
#pragma unroll
    for (int j = 0; j < 2; ++j) {
        int s   = tid + j * 256;
        int row = s >> 2;                         // 0..127
        int k8  = (s & 3) * 8;
        dstA[j] = frag_index(row, k8);
        int gm  = min(blockM + row, M - 1);       // clamp: address-safe
        long ar = rowmap ? (long)rowmap[gm] : (long)gm;
        aoff[j] = ar * Kp + k8;
    }
    const int  rowB = tid >> 2;                   // 0..63
    const int  k8B  = (tid & 3) * 8;
    const int  dstB = frag_index(rowB, k8B);
    const long boff = (long)min(blockN + rowB, N - 1) * Kp + k8B;

    v8bf ra0, ra1, rb;
    auto load_tile = [&](int kt) {                // globals -> registers
        const long k0 = (long)kt * BK;
        ra0 = *reinterpret_cast<const v8bf*>(A + aoff[0] + k0);
        ra1 = *reinterpret_cast<const v8bf*>(A + aoff[1] + k0);
        rb  = *reinterpret_cast<const v8bf*>(W + boff + k0);
    };
    auto store_tile = [&](int buf) {              // registers -> LDS (frag layout)
        *reinterpret_cast<v8bf*>(&sA[buf][dstA[0]]) = ra0;
        *reinterpret_cast<v8bf*>(&sA[buf][dstA[1]]) = ra1;
        *reinterpret_cast<v8bf*>(&sB[buf][dstB])    = rb;
    };

    v8f acc00 = {}, acc01 = {}, acc10 = {}, acc11 = {};
    const int ktiles = Kp / BK;                   // exact (Kp % 32 == 0)
    const int mt0 = wm * 2, mt1 = wm * 2 + 1;
    const int nt0 = wn * 2, nt1 = wn * 2 + 1;

    load_tile(0);
    store_tile(0);
    __syncthreads();

    for (int kt = 0; kt < ktiles; ++kt) {
        const int  cur  = kt & 1;
        const bool more = (kt + 1) < ktiles;
        if (more) load_tile(kt + 1);              // VMEM overlaps the WMMAs below

        v16bf a0 = ld_frag(&sA[cur][(mt0 * 32 + lane) * 16]);
        v16bf a1 = ld_frag(&sA[cur][(mt1 * 32 + lane) * 16]);
        v16bf b0 = ld_frag(&sB[cur][(nt0 * 32 + lane) * 16]);
        v16bf b1 = ld_frag(&sB[cur][(nt1 * 32 + lane) * 16]);
        acc00 = __builtin_amdgcn_wmma_f32_16x16x32_bf16(false, a0, false, b0,
                                                        (short)0, acc00, false, false);
        acc01 = __builtin_amdgcn_wmma_f32_16x16x32_bf16(false, a0, false, b1,
                                                        (short)0, acc01, false, false);
        acc10 = __builtin_amdgcn_wmma_f32_16x16x32_bf16(false, a1, false, b0,
                                                        (short)0, acc10, false, false);
        acc11 = __builtin_amdgcn_wmma_f32_16x16x32_bf16(false, a1, false, b1,
                                                        (short)0, acc11, false, false);
        if (more) store_tile(1 - cur);            // fill the other buffer
        __syncthreads();                          // single barrier per K-step
    }

    // --- store: lane = col N within subtile, VGPR r = row (lanes 16..31: +8) ---
    const int lidx  = lane & 15;
    const int rhalf = (lane >> 4) ? 8 : 0;
    const int gn0 = blockN + nt0 * 16 + lidx;
    const int gn1 = blockN + nt1 * 16 + lidx;
    float bv0 = (bias != nullptr && gn0 < N) ? bias[gn0] : 0.0f;
    float bv1 = (bias != nullptr && gn1 < N) ? bias[gn1] : 0.0f;
#pragma unroll
    for (int mi = 0; mi < 2; ++mi) {
        const v8f& c0 = mi ? acc10 : acc00;
        const v8f& c1 = mi ? acc11 : acc01;
        int gmb = blockM + (wm * 2 + mi) * 16 + rhalf;
#pragma unroll
        for (int r = 0; r < 8; ++r) {
            int gm = gmb + r;
            if (gm < M) {
                if (gn0 < N) {
                    float v = c0[r] + bv0;
                    if (RELU) v = fmaxf(v, 0.0f);
                    C[(long)gm * N + gn0] = v;
                }
                if (gn1 < N) {
                    float v = c1[r] + bv1;
                    if (RELU) v = fmaxf(v, 0.0f);
                    C[(long)gm * N + gn1] = v;
                }
            }
        }
    }
}

// agg[dst] += w_e * x[src] over all edges; consecutive threads cover one edge's features.
__global__ __launch_bounds__(256)
void scatter_edges(const float* __restrict__ x, const int* __restrict__ edge_index,
                   const float* __restrict__ ew, float* __restrict__ agg, int F)
{
    const int chunks = F >> 2;                       // F is a multiple of 4
    long idx = (long)blockIdx.x * blockDim.x + threadIdx.x;
    if (idx >= (long)NEDGES * chunks) return;
    int c = (int)(idx % chunks);
    int e = (int)(idx / chunks);
    int s = edge_index[e];
    int d = edge_index[NEDGES + e];
    float w = ew[e];
    const float4 v = *reinterpret_cast<const float4*>(x + (long)s * F + c * 4);
    float* o = agg + (long)d * F + c * 4;
    atomicAdd(o + 0, w * v.x);
    atomicAdd(o + 1, w * v.y);
    atomicAdd(o + 2, w * v.z);
    atomicAdd(o + 3, w * v.w);
}

__global__ void gather_y(const int* __restrict__ mask, const int* __restrict__ y,
                         float* __restrict__ outy)
{
    int i = blockIdx.x * blockDim.x + threadIdx.x;
    if (i < NMASK) outy[i] = (float)y[mask[i]];
}

static inline void cast_pad(const float* in, __bf16* out, int rows, int K, int Kp,
                            int ldout, hipStream_t stream) {
    long total = (long)rows * (Kp >> 2);
    cast_pad_bf16<<<(unsigned)((total + 255) / 256), 256, 0, stream>>>(
        in, out, rows, K, Kp, ldout);
}

extern "C" void kernel_launch(void* const* d_in, const int* in_sizes, int n_in,
                              void* d_out, int out_size, void* d_ws, size_t ws_size,
                              hipStream_t stream) {
    const float* doc_features  = (const float*)d_in[0];
    const float* word_features = (const float*)d_in[1];
    const int*   edge_index    = (const int*)  d_in[2];
    const float* edge_weight   = (const float*)d_in[3];
    const int*   mask          = (const int*)  d_in[4];
    const int*   y             = (const int*)  d_in[5];
    const float* lin_w         = (const float*)d_in[6];
    const float* lin_b         = (const float*)d_in[7];
    const float* w1_rel        = (const float*)d_in[8];
    const float* b1            = (const float*)d_in[9];
    const float* w1_root       = (const float*)d_in[10];
    const float* w2_rel        = (const float*)d_in[11];
    const float* b2            = (const float*)d_in[12];
    const float* w2_root       = (const float*)d_in[13];
    float* out = (float*)d_out;

    // --- workspace layout ---
    char* p = (char*)d_ws;
    auto alloc_f = [&](size_t n) { float*  r = (float*) p; p += n * sizeof(float);  return r; };
    auto alloc_b = [&](size_t n) { __bf16* r = (__bf16*)p; p += n * sizeof(__bf16); return r; };
    // fp32 intermediates
    float* x    = alloc_f((size_t)NNODES * WORD_DIM);     // 36 MB
    float* agg1 = alloc_f((size_t)NNODES * WORD_DIM);     // 36 MB
    float* h1   = alloc_f((size_t)NNODES * OUT_DIM);      // 92 MB
    float* agg2 = alloc_f((size_t)NNODES * OUT_DIM);      // 92 MB
    // bf16 operands (K padded / K-concatenated)
    __bf16* dfb   = alloc_b((size_t)N_DOCS * DOC_DIM_P);    // 80 MB
    __bf16* lwb   = alloc_b((size_t)WORD_DIM * DOC_DIM_P);  // 6 MB
    __bf16* cat1  = alloc_b((size_t)NNODES * K_CAT1);       // [agg1 | x]  38 MB
    __bf16* w1cat = alloc_b((size_t)OUT_DIM * K_CAT1);      // [w1_rel | w1_root]
    __bf16* cat2  = alloc_b((size_t)NNODES * K_CAT2);       // [agg2 | h1] 92 MB
    __bf16* w2cat = alloc_b((size_t)OUT_DIM * K_CAT2);      // [w2_rel | w2_root]

    // 1) doc = doc_features @ lin_w.T + lin_b  -> x[0:4000]
    cast_pad(doc_features, dfb, N_DOCS, DOC_DIM, DOC_DIM_P, DOC_DIM_P, stream);
    cast_pad(lin_w, lwb, WORD_DIM, DOC_DIM, DOC_DIM_P, DOC_DIM_P, stream);
    {
        dim3 grid((WORD_DIM + 63) / 64, (N_DOCS + 127) / 128);
        gemm_bf16_wmma<false><<<grid, 256, 0, stream>>>(
            dfb, lwb, lin_b, nullptr, x, N_DOCS, WORD_DIM, DOC_DIM_P);
    }
    // x[4000:] = word_features
    hipMemcpyAsync(x + (size_t)N_DOCS * WORD_DIM, word_features,
                   (size_t)N_WORDS * WORD_DIM * sizeof(float),
                   hipMemcpyDeviceToDevice, stream);

    // 2) agg1 = segment_sum(edge_weight * x[src]) over dst
    hipMemsetAsync(agg1, 0, (size_t)NNODES * WORD_DIM * sizeof(float), stream);
    {
        long total = (long)NEDGES * (WORD_DIM / 4);
        scatter_edges<<<(unsigned)((total + 255) / 256), 256, 0, stream>>>(
            x, edge_index, edge_weight, agg1, WORD_DIM);
    }

    // 3) h1 = relu([agg1|x] @ [w1_rel|w1_root]^T + b1)  (K-concatenated GEMM)
    cast_pad(agg1, cat1, NNODES, WORD_DIM, WORD_DIM_P, K_CAT1, stream);
    cast_pad(x, cat1 + WORD_DIM_P, NNODES, WORD_DIM, WORD_DIM_P, K_CAT1, stream);
    cast_pad(w1_rel, w1cat, OUT_DIM, WORD_DIM, WORD_DIM_P, K_CAT1, stream);
    cast_pad(w1_root, w1cat + WORD_DIM_P, OUT_DIM, WORD_DIM, WORD_DIM_P, K_CAT1, stream);
    {
        dim3 grid((OUT_DIM + 63) / 64, (NNODES + 127) / 128);
        gemm_bf16_wmma<true><<<grid, 256, 0, stream>>>(
            cat1, w1cat, b1, nullptr, h1, NNODES, OUT_DIM, K_CAT1);
    }

    // 4) agg2 = segment_sum(edge_weight * h1[src]) over dst
    hipMemsetAsync(agg2, 0, (size_t)NNODES * OUT_DIM * sizeof(float), stream);
    {
        long total = (long)NEDGES * (OUT_DIM / 4);
        scatter_edges<<<(unsigned)((total + 255) / 256), 256, 0, stream>>>(
            h1, edge_index, edge_weight, agg2, OUT_DIM);
    }

    // 5) out[0:4096,:] = ([agg2|h1] @ [w2_rel|w2_root]^T + b2)[mask]
    //    masked GEMM via rowmap: only the 4096 selected rows are computed.
    cast_pad(agg2, cat2, NNODES, OUT_DIM, OUT_DIM, K_CAT2, stream);
    cast_pad(h1, cat2 + OUT_DIM, NNODES, OUT_DIM, OUT_DIM, K_CAT2, stream);
    cast_pad(w2_rel, w2cat, OUT_DIM, OUT_DIM, OUT_DIM, K_CAT2, stream);
    cast_pad(w2_root, w2cat + OUT_DIM, OUT_DIM, OUT_DIM, OUT_DIM, K_CAT2, stream);
    {
        dim3 grid((OUT_DIM + 63) / 64, (NMASK + 127) / 128);
        gemm_bf16_wmma<false><<<grid, 256, 0, stream>>>(
            cat2, w2cat, b2, mask, out, NMASK, OUT_DIM, K_CAT2);
    }

    // 6) out_y = y[mask]
    gather_y<<<(NMASK + 255) / 256, 256, 0, stream>>>(
        mask, y, out + (size_t)NMASK * OUT_DIM);
}
